// RecurrentAutoEncoder_23046794510974
// MI455X (gfx1250) — matrix operand order repaired and checked by reading, not compiled, not run on hardware
//
#include <hip/hip_runtime.h>
#include <math.h>

#define BB 64
#define TT 128
#define II 512
#define HH 1024
#define G4 4096  // 4*H

typedef __attribute__((ext_vector_type(16))) __bf16 bf16x16;
typedef __attribute__((ext_vector_type(8)))  float  f32x8;
typedef __attribute__((ext_vector_type(4)))  unsigned int u32x4;

union FragU { bf16x16 v; u32x4 q[2]; };

// A fragment (16x32 bf16, row-major source).
// ISA layout: lanes 0-15 -> row M=lane, K = 0..7 (v0..v3) and 16..23 (v4..v7);
// lanes 16-31 -> same rows, K = 8..15 and 24..31.
__device__ __forceinline__ bf16x16 load_a_frag(const __bf16* base, int ld,
                                               int m0, int k0, int lane) {
  int m  = m0 + (lane & 15);
  int kh = lane >> 4;
  const __bf16* p = base + (size_t)m * ld + (k0 + kh * 8);
  FragU f;
  f.q[0] = *(const u32x4*)(p);        // K = base .. base+7
  f.q[1] = *(const u32x4*)(p + 16);   // K = base+16 .. base+23
  return f.v;
}

// B fragment (32x16 bf16). Column n of B == row n of W ([N,K] row-major).
// ISA layout: lanes 0-15 -> N=lane, K=0..7 and 16..23; lanes 16-31 -> N=lane-16, K=8..15 and 24..31.
__device__ __forceinline__ bf16x16 load_b_frag(const __bf16* wbase, int ld,
                                               int n0, int k0, int lane) {
  int n  = n0 + (lane & 15);
  int kh = lane >> 4;
  const __bf16* p = wbase + (size_t)n * ld + (k0 + kh * 8);
  FragU f;
  f.q[0] = *(const u32x4*)(p);        // K = base .. base+7
  f.q[1] = *(const u32x4*)(p + 16);   // K = base+16 .. base+23
  return f.v;
}

__device__ __forceinline__ f32x8 wmma_bf16(bf16x16 a, bf16x16 b, f32x8 c) {
#if defined(__gfx1250__)
  return __builtin_amdgcn_wmma_f32_16x16x32_bf16(false, a, false, b,
                                                 (short)0, c, false, false);
#else
  return c;
#endif
}

__device__ __forceinline__ float sigf(float x) {
  return 1.f / (1.f + __expf(-x));
}

// ---------------------------------------------------------------------------
// C[M,N](bf16) = A[M,K](bf16) @ W[N,K](bf16)^T + bias[N](f32)
// Block: 256 thr (8 waves), tile 128(M) x 128(N); wave tile 32x64 (2x4 frags).
// Grid: (N/128, M/128)
// ---------------------------------------------------------------------------
__global__ __launch_bounds__(256) void gemm_bias_bf16(
    const __bf16* __restrict__ A, const __bf16* __restrict__ W,
    const float* __restrict__ bias, __bf16* __restrict__ out,
    int N, int K) {
  int lane = threadIdx.x & 31;
  int wave = threadIdx.x >> 5;
  int m0 = blockIdx.y * 128 + (wave >> 1) * 32;
  int n0 = blockIdx.x * 128 + (wave & 1) * 64;

  f32x8 acc[2][4] = {};
  for (int k = 0; k < K; k += 32) {
    bf16x16 afr[2], bfr[4];
    afr[0] = load_a_frag(A, K, m0,      k, lane);
    afr[1] = load_a_frag(A, K, m0 + 16, k, lane);
    bfr[0] = load_b_frag(W, K, n0,      k, lane);
    bfr[1] = load_b_frag(W, K, n0 + 16, k, lane);
    bfr[2] = load_b_frag(W, K, n0 + 32, k, lane);
    bfr[3] = load_b_frag(W, K, n0 + 48, k, lane);
#pragma unroll
    for (int i = 0; i < 2; ++i)
#pragma unroll
      for (int j = 0; j < 4; ++j)
        acc[i][j] = wmma_bf16(afr[i], bfr[j], acc[i][j]);
  }

  int mo  = (lane >> 4) * 8;
  int nlo = lane & 15;
#pragma unroll
  for (int i = 0; i < 2; ++i)
#pragma unroll
    for (int j = 0; j < 4; ++j) {
      int n = n0 + j * 16 + nlo;
      float bv = bias[n];
#pragma unroll
      for (int r = 0; r < 8; ++r) {
        int m = m0 + i * 16 + r + mo;
        out[(size_t)m * N + n] = (__bf16)(acc[i][j][r] + bv);
      }
    }
}

// ---------------------------------------------------------------------------
// One LSTM time step, fused: gates = gpre + h_prev @ Whh^T, then cell update.
// Grid: 16 blocks (each owns 64 hidden columns across all 4 gates), 256 thr.
// Wave w: gate = w/2, 32-col half = (w&1). Accum 64(M=B) x 32 per wave.
// ---------------------------------------------------------------------------
__global__ __launch_bounds__(256) void lstm_step(
    const __bf16* __restrict__ Whh,    // [4H][H] bf16
    const __bf16* __restrict__ gpre,   // [B][4H] bf16 (this timestep)
    const __bf16* __restrict__ h_prev, // [B][H]  bf16
    float* __restrict__ c_state,       // [B][H]  f32 (in/out)
    __bf16* __restrict__ h_out) {      // [B][H]  bf16
  __shared__ float g_lds[64][256];

  int lane = threadIdx.x & 31;
  int wave = threadIdx.x >> 5;
  int gate = wave >> 1;
  int sub  = (wave & 1) * 32;
  int n0   = blockIdx.x * 64;
  int colbase = gate * HH + n0 + sub;  // row of Whh / column of gates

  f32x8 acc[4][2] = {};
  for (int k = 0; k < HH; k += 32) {
    bf16x16 afr[4], bfr[2];
    afr[0] = load_a_frag(h_prev, HH, 0,  k, lane);
    afr[1] = load_a_frag(h_prev, HH, 16, k, lane);
    afr[2] = load_a_frag(h_prev, HH, 32, k, lane);
    afr[3] = load_a_frag(h_prev, HH, 48, k, lane);
    bfr[0] = load_b_frag(Whh, HH, colbase,      k, lane);
    bfr[1] = load_b_frag(Whh, HH, colbase + 16, k, lane);
#pragma unroll
    for (int i = 0; i < 4; ++i)
#pragma unroll
      for (int j = 0; j < 2; ++j)
        acc[i][j] = wmma_bf16(afr[i], bfr[j], acc[i][j]);
  }

  int mo  = (lane >> 4) * 8;
  int nlo = lane & 15;
#pragma unroll
  for (int i = 0; i < 4; ++i)
#pragma unroll
    for (int j = 0; j < 2; ++j)
#pragma unroll
      for (int r = 0; r < 8; ++r)
        g_lds[i * 16 + r + mo][gate * 64 + sub + j * 16 + nlo] = acc[i][j][r];
  __syncthreads();

  // Cell update: thread -> column jc, 16 batch rows each.
  int jc  = threadIdx.x & 63;
  int b0r = (threadIdx.x >> 6) * 16;
  int n   = n0 + jc;
  for (int bb = 0; bb < 16; ++bb) {
    int b = b0r + bb;
    size_t gb = (size_t)b * G4 + n;
    float ig = g_lds[b][jc]        + (float)gpre[gb];
    float fg = g_lds[b][64 + jc]   + (float)gpre[gb + HH];
    float gg = g_lds[b][128 + jc]  + (float)gpre[gb + 2 * HH];
    float og = g_lds[b][192 + jc]  + (float)gpre[gb + 3 * HH];
    size_t ci = (size_t)b * HH + n;
    float cn = sigf(fg) * c_state[ci] + sigf(ig) * tanhf(gg);
    c_state[ci] = cn;
    h_out[ci] = (__bf16)(sigf(og) * tanhf(cn));
  }
}

// ---------------------------------------------------------------------------
// out[B,T,I](f32) = sigmoid(ys[T*B,H] @ fcW[I,H]^T + fcb). Grid (I/128, TB/128).
// ---------------------------------------------------------------------------
__global__ __launch_bounds__(256) void fc_sigmoid(
    const __bf16* __restrict__ A, const __bf16* __restrict__ W,
    const float* __restrict__ bias, float* __restrict__ out) {
  int lane = threadIdx.x & 31;
  int wave = threadIdx.x >> 5;
  int m0 = blockIdx.y * 128 + (wave >> 1) * 32;
  int n0 = blockIdx.x * 128 + (wave & 1) * 64;

  f32x8 acc[2][4] = {};
  for (int k = 0; k < HH; k += 32) {
    bf16x16 afr[2], bfr[4];
    afr[0] = load_a_frag(A, HH, m0,      k, lane);
    afr[1] = load_a_frag(A, HH, m0 + 16, k, lane);
    bfr[0] = load_b_frag(W, HH, n0,      k, lane);
    bfr[1] = load_b_frag(W, HH, n0 + 16, k, lane);
    bfr[2] = load_b_frag(W, HH, n0 + 32, k, lane);
    bfr[3] = load_b_frag(W, HH, n0 + 48, k, lane);
#pragma unroll
    for (int i = 0; i < 2; ++i)
#pragma unroll
      for (int j = 0; j < 4; ++j)
        acc[i][j] = wmma_bf16(afr[i], bfr[j], acc[i][j]);
  }

  int mo  = (lane >> 4) * 8;
  int nlo = lane & 15;
#pragma unroll
  for (int i = 0; i < 2; ++i)
#pragma unroll
    for (int j = 0; j < 4; ++j) {
      int n = n0 + j * 16 + nlo;
      float bv = bias[n];
#pragma unroll
      for (int r = 0; r < 8; ++r) {
        int m = m0 + i * 16 + r + mo;  // m = t*B + b
        int b = m & (BB - 1);
        int t = m >> 6;
        out[((size_t)b * TT + t) * II + n] = sigf(acc[i][j][r] + bv);
      }
    }
}

// ------------------------- small elementwise kernels ------------------------
__global__ void cvt_bf16(const float* __restrict__ s, __bf16* __restrict__ d,
                         int n) {
  int i = blockIdx.x * blockDim.x + threadIdx.x;
  if (i < n) d[i] = (__bf16)s[i];
}

// dst[(t*B+b)*I + i] = (t < shift) ? 0 : x[(b*T + t-shift)*I + i]
__global__ void xpose_cvt(const float* __restrict__ x, __bf16* __restrict__ dst,
                          int shift) {
  size_t idx = (size_t)blockIdx.x * blockDim.x + threadIdx.x;
  int i = (int)(idx % II);
  size_t r = idx / II;
  int b = (int)(r % BB);
  int t = (int)(r / BB);
  float v = 0.f;
  if (t >= shift) v = x[((size_t)b * TT + (t - shift)) * II + i];
  dst[idx] = (__bf16)v;
}

__global__ void tanh_bf16(const __bf16* __restrict__ s, __bf16* __restrict__ d,
                          int n) {
  int i = blockIdx.x * blockDim.x + threadIdx.x;
  if (i < n) d[i] = (__bf16)tanhf((float)s[i]);
}

// ---------------------------------------------------------------------------
extern "C" void kernel_launch(void* const* d_in, const int* in_sizes, int n_in,
                              void* d_out, int out_size, void* d_ws,
                              size_t ws_size, hipStream_t stream) {
  (void)in_sizes; (void)n_in; (void)out_size; (void)ws_size;
  const float* x      = (const float*)d_in[0];
  const float* dec_c0 = (const float*)d_in[1];
  const float* eWih0  = (const float*)d_in[2];
  const float* eWhh0  = (const float*)d_in[3];
  const float* eb0    = (const float*)d_in[4];
  const float* eWih1  = (const float*)d_in[5];
  const float* eWhh1  = (const float*)d_in[6];
  const float* eb1    = (const float*)d_in[7];
  const float* dWih0  = (const float*)d_in[8];
  const float* dWhh0  = (const float*)d_in[9];
  const float* db0    = (const float*)d_in[10];
  const float* dWih1  = (const float*)d_in[11];
  const float* dWhh1  = (const float*)d_in[12];
  const float* db1    = (const float*)d_in[13];
  const float* fcW    = (const float*)d_in[14];
  const float* fcb    = (const float*)d_in[15];
  float* out = (float*)d_out;

  char* base = (char*)d_ws;
  size_t off = 0;
  auto take = [&](size_t bytes) -> char* {
    char* p = base + off;
    off += (bytes + 255) & ~(size_t)255;
    return p;
  };

  __bf16* xs     = (__bf16*)take((size_t)TT * BB * II * 2);
  __bf16* tok    = (__bf16*)take((size_t)TT * BB * II * 2);
  __bf16* eWih0b = (__bf16*)take((size_t)G4 * II * 2);
  __bf16* eWhh0b = (__bf16*)take((size_t)G4 * HH * 2);
  __bf16* eWih1b = (__bf16*)take((size_t)G4 * HH * 2);
  __bf16* eWhh1b = (__bf16*)take((size_t)G4 * HH * 2);
  __bf16* dWih0b = (__bf16*)take((size_t)G4 * II * 2);
  __bf16* dWhh0b = (__bf16*)take((size_t)G4 * HH * 2);
  __bf16* dWih1b = (__bf16*)take((size_t)G4 * HH * 2);
  __bf16* dWhh1b = (__bf16*)take((size_t)G4 * HH * 2);
  __bf16* fcWb   = (__bf16*)take((size_t)II * HH * 2);
  __bf16* gpre   = (__bf16*)take((size_t)TT * BB * G4 * 2);  // 64 MB
  __bf16* seq0   = (__bf16*)take((size_t)TT * BB * HH * 2);
  __bf16* seq1   = (__bf16*)take((size_t)TT * BB * HH * 2);
  __bf16* hzero  = (__bf16*)take((size_t)BB * HH * 2);
  __bf16* hi0    = (__bf16*)take((size_t)BB * HH * 2);
  __bf16* hi1    = (__bf16*)take((size_t)BB * HH * 2);
  float*  cst    = (float*)take((size_t)BB * HH * 4);

  auto cvt = [&](const float* s, __bf16* d, size_t n) {
    cvt_bf16<<<(unsigned)((n + 255) / 256), 256, 0, stream>>>(s, d, (int)n);
  };

  // ---- conversions (bf16, L2-resident weights) ----
  xpose_cvt<<<TT * BB * II / 256, 256, 0, stream>>>(x, xs, 0);
  xpose_cvt<<<TT * BB * II / 256, 256, 0, stream>>>(x, tok, 1);
  cvt(eWih0, eWih0b, (size_t)G4 * II);
  cvt(eWhh0, eWhh0b, (size_t)G4 * HH);
  cvt(eWih1, eWih1b, (size_t)G4 * HH);
  cvt(eWhh1, eWhh1b, (size_t)G4 * HH);
  cvt(dWih0, dWih0b, (size_t)G4 * II);
  cvt(dWhh0, dWhh0b, (size_t)G4 * HH);
  cvt(dWih1, dWih1b, (size_t)G4 * HH);
  cvt(dWhh1, dWhh1b, (size_t)G4 * HH);
  cvt(fcW, fcWb, (size_t)II * HH);
  hipMemsetAsync(hzero, 0, (size_t)BB * HH * 2, stream);

  auto run_layer = [&](const __bf16* inseq, int Kin, const __bf16* Wihb,
                       const float* bias, const __bf16* Whhb,
                       const __bf16* h0, __bf16* outseq) {
    // Parallel pre-activation GEMM over all T*B rows (throughput-bound).
    gemm_bias_bf16<<<dim3(G4 / 128, TT * BB / 128), 256, 0, stream>>>(
        inseq, Wihb, bias, gpre, G4, Kin);
    // Sequential recurrence (latency-bound critical path).
    for (int t = 0; t < TT; ++t) {
      const __bf16* hp = (t == 0) ? h0 : outseq + (size_t)(t - 1) * BB * HH;
      lstm_step<<<HH / 64, 256, 0, stream>>>(
          Whhb, gpre + (size_t)t * BB * G4, hp, cst,
          outseq + (size_t)t * BB * HH);
    }
  };

  // ---- Encoder ----
  hipMemsetAsync(cst, 0, (size_t)BB * HH * 4, stream);
  run_layer(xs, II, eWih0b, eb0, eWhh0b, hzero, seq0);
  tanh_bf16<<<BB * HH / 256, 256, 0, stream>>>(
      seq0 + (size_t)(TT - 1) * BB * HH, hi0, BB * HH);

  hipMemsetAsync(cst, 0, (size_t)BB * HH * 4, stream);
  run_layer(seq0, HH, eWih1b, eb1, eWhh1b, hzero, seq1);
  tanh_bf16<<<BB * HH / 256, 256, 0, stream>>>(
      seq1 + (size_t)(TT - 1) * BB * HH, hi1, BB * HH);

  // ---- Decoder (teacher forcing; h0 = tanh(enc h_T), c0 = dec_c0) ----
  hipMemcpyAsync(cst, dec_c0, (size_t)BB * HH * 4,
                 hipMemcpyDeviceToDevice, stream);
  run_layer(tok, II, dWih0b, db0, dWhh0b, hi0, seq0);

  hipMemcpyAsync(cst, dec_c0 + (size_t)BB * HH, (size_t)BB * HH * 4,
                 hipMemcpyDeviceToDevice, stream);
  run_layer(seq0, HH, dWih1b, db1, dWhh1b, hi1, seq1);

  // ---- FC + sigmoid -> [B,T,I] f32 ----
  fc_sigmoid<<<dim3(II / 128, TT * BB / 128), 256, 0, stream>>>(
      seq1, fcWb, fcb, out);
}
